// BitLinear_17051020165443
// MI455X (gfx1250) — matrix operand order repaired and checked by reading, not compiled, not run on hardware
//
#include <hip/hip_runtime.h>

// BitLinear: y[t,o] = (sum_k x[t,k] * sign(W[o,k])) * scale[o]
// M=8192 (tokens), K=4096 (in_f), N=16384 (out_f).
// Compute-bound GEMM -> drive V_WMMA_F32_16X16X32_F16 from LDS tiles.

typedef __attribute__((ext_vector_type(16))) _Float16 v16h;
typedef __attribute__((ext_vector_type(8)))  _Float16 v8h;
typedef __attribute__((ext_vector_type(8)))  float    v8f;
typedef __attribute__((ext_vector_type(4)))  float    v4f;

#define TOKENS 8192
#define IN_F   4096
#define OUT_F  16384

#define BM 128
#define BN 128
#define BK 64
#define SA 72               // padded LDS stride in halves (144 B, 16B-aligned)
#define NCH (IN_F / BK)     // 64 K-chunks

__device__ __forceinline__ _Float16 sgn_h(float w) {
    float s = (w > 0.0f) ? 1.0f : ((w < 0.0f) ? -1.0f : 0.0f);
    return (_Float16)s;
}

__global__ __launch_bounds__(256)
void bitlinear_wmma_f16(const float* __restrict__ X,
                        const float* __restrict__ W,
                        const float* __restrict__ scale,
                        float* __restrict__ Y)
{
    __shared__ __attribute__((aligned(16))) _Float16 As[2][BM * SA];
    __shared__ __attribute__((aligned(16))) _Float16 Bs[2][BN * SA];

    const int tid  = threadIdx.x;
    const int lane = tid & 31;
    const int wave = tid >> 5;
    const int wm   = wave & 3;        // 4 waves over M: 32 rows each
    const int wn   = wave >> 2;       // 2 waves over N: 64 cols each
    const int tileM = blockIdx.y * BM;
    const int tileN = blockIdx.x * BN;

    // --- global -> register staging assignment (256 thr: 2 per row, 32 f32 each)
    const int srow = tid >> 1;              // 0..127
    const int scol = (tid & 1) * 32;        // 0 or 32 within the 64-wide K chunk

    const float* gX = X + (size_t)(tileM + srow) * IN_F + scol;
    const float* gW = W + (size_t)(tileN + srow) * IN_F + scol;

    v4f ra[8], rb[8];

    auto load_regs = [&](int kc) {
        const float* pa = gX + (size_t)kc * BK;
        const float* pw = gW + (size_t)kc * BK;
#pragma unroll
        for (int j = 0; j < 8; ++j) {
            ra[j] = *(const v4f*)(pa + j * 4);   // global_load_b128
            rb[j] = *(const v4f*)(pw + j * 4);
        }
    };

    auto store_lds = [&](int buf) {
        _Float16* da = &As[buf][srow * SA + scol];
        _Float16* db = &Bs[buf][srow * SA + scol];
#pragma unroll
        for (int j = 0; j < 4; ++j) {
            v4f a0 = ra[2 * j], a1 = ra[2 * j + 1];
            v8h ha;
            ha[0] = (_Float16)a0[0]; ha[1] = (_Float16)a0[1];
            ha[2] = (_Float16)a0[2]; ha[3] = (_Float16)a0[3];
            ha[4] = (_Float16)a1[0]; ha[5] = (_Float16)a1[1];
            ha[6] = (_Float16)a1[2]; ha[7] = (_Float16)a1[3];
            *(v8h*)(da + j * 8) = ha;            // ds_store_b128

            v4f w0 = rb[2 * j], w1 = rb[2 * j + 1];
            v8h hb;
            hb[0] = sgn_h(w0[0]); hb[1] = sgn_h(w0[1]);
            hb[2] = sgn_h(w0[2]); hb[3] = sgn_h(w0[3]);
            hb[4] = sgn_h(w1[0]); hb[5] = sgn_h(w1[1]);
            hb[6] = sgn_h(w1[2]); hb[7] = sgn_h(w1[3]);
            *(v8h*)(db + j * 8) = hb;
        }
    };

    v8f acc[2][4];
#pragma unroll
    for (int m = 0; m < 2; ++m)
#pragma unroll
        for (int n = 0; n < 4; ++n)
            acc[m][n] = 0;

    const int lm = lane & 15;
    const int hi = lane >> 4;

    auto compute = [&](int buf) {
        const _Float16* A0 = As[buf];
        const _Float16* B0 = Bs[buf];
#pragma unroll
        for (int ks = 0; ks < 2; ++ks) {
            // A 16-bit 16x32 layout: lanes0-15 K 0-7 & 16-23; lanes16-31 K 8-15 & 24-31
            const int kA = ks * 32 + hi * 8;
            // B 16-bit 32x16 layout: lanes0-15 K 0-7 & 16-23; lanes16-31 K 8-15 & 24-31
            const int kB = ks * 32 + hi * 8;

            v16h af[2], bf[4];
#pragma unroll
            for (int m = 0; m < 2; ++m) {
                const _Float16* p = A0 + (wm * 32 + m * 16 + lm) * SA + kA;
                v8h lo = *(const v8h*)p;
                v8h hv = *(const v8h*)(p + 16);
                af[m] = __builtin_shufflevector(lo, hv,
                        0,1,2,3,4,5,6,7,8,9,10,11,12,13,14,15);
            }
#pragma unroll
            for (int n = 0; n < 4; ++n) {
                const _Float16* p = B0 + (wn * 64 + n * 16 + lm) * SA + kB;
                v8h lo = *(const v8h*)p;
                v8h hv = *(const v8h*)(p + 16);
                bf[n] = __builtin_shufflevector(lo, hv,
                        0,1,2,3,4,5,6,7,8,9,10,11,12,13,14,15);
            }
#pragma unroll
            for (int m = 0; m < 2; ++m)
#pragma unroll
                for (int n = 0; n < 4; ++n)
                    acc[m][n] = __builtin_amdgcn_wmma_f32_16x16x32_f16(
                        /*neg_a=*/false, af[m], /*neg_b=*/false, bf[n],
                        /*c_mod=*/(short)0, acc[m][n],
                        /*reuse_a=*/false, /*reuse_b=*/false);
        }
    };

    // ---- pipeline: regs(kc+1) in flight while WMMA consumes buf(kc) ----
    load_regs(0);
    store_lds(0);
    __syncthreads();

    for (int kc = 0; kc < NCH; ++kc) {
        const int cur = kc & 1;
        if (kc + 1 < NCH) load_regs(kc + 1);   // global loads overlap compute
        compute(cur);
        if (kc + 1 < NCH) {
            __syncthreads();                   // readers of buf cur^1 done
            store_lds(cur ^ 1);
            __syncthreads();                   // buf cur^1 ready
        }
    }

    // ---- epilogue: C layout lane=N, VGPR r = M row; apply per-N scale ----
    const int mh = hi * 8;
#pragma unroll
    for (int n = 0; n < 4; ++n) {
        const int gN = tileN + wn * 64 + n * 16 + lm;
        const float sc = scale[gN];
#pragma unroll
        for (int m = 0; m < 2; ++m) {
            const int gMb = tileM + wm * 32 + m * 16 + mh;
#pragma unroll
            for (int r = 0; r < 8; ++r)
                Y[(size_t)(gMb + r) * OUT_F + gN] = acc[m][n][r] * sc;
        }
    }
}

extern "C" void kernel_launch(void* const* d_in, const int* in_sizes, int n_in,
                              void* d_out, int out_size, void* d_ws, size_t ws_size,
                              hipStream_t stream) {
    (void)in_sizes; (void)n_in; (void)out_size; (void)d_ws; (void)ws_size;
    const float* X = (const float*)d_in[0];      // [TOKENS, IN_F]
    const float* W = (const float*)d_in[1];      // [OUT_F, IN_F]
    const float* S = (const float*)d_in[2];      // [OUT_F]
    float* Y = (float*)d_out;                    // [TOKENS, OUT_F]

    dim3 grid(OUT_F / BN, TOKENS / BM);          // 128 x 64 = 8192 WGs
    bitlinear_wmma_f16<<<grid, 256, 0, stream>>>(X, W, S, Y);
}
